// SwinTransformerBlock_3882650435595
// MI455X (gfx1250) — matrix operand order repaired and checked by reading, not compiled, not run on hardware
//
#include <hip/hip_runtime.h>
#include <hip/hip_bf16.h>

// ---------------------------------------------------------------------------
// Fused Swin Transformer block for MI455X (gfx1250, wave32, WMMA).
// One workgroup (256 threads = 8 waves) processes one 7x7 window entirely in
// LDS; all GEMMs run on v_wmma_f32_16x16x32_bf16 with fp32 accumulation.
// Weights are pre-converted fp32 -> bf16 into d_ws (stay hot in 192MB L2).
// ---------------------------------------------------------------------------

typedef __attribute__((ext_vector_type(16))) __bf16       v16bf;
typedef __attribute__((ext_vector_type(8)))  float        v8f;
typedef __attribute__((ext_vector_type(4)))  unsigned int u32x4;

union Frag  { v16bf h; u32x4 u[2]; };
union Pack8 { u32x4 u; __bf16 e[8]; };      // 8 bf16 -> one b128 store

#define CC   256     // channels
#define TT   49      // tokens per window (7x7)
#define HD   32      // head dim
#define WSZ  7
#define HP   112     // H == W (112 % 7 == 0 -> no pad)
#define SSH  3       // shift

// ---- LDS layout (bytes); all offsets 16B aligned ----
#define XW_STRIDE 260                    // fp32, pad for banks
#define XN_STRIDE 264                    // bf16 token-major buffers
#define VT_STRIDE 72                     // bf16 V^T rows (64 tokens + pad)
#define P_STRIDE  72                     // bf16 prob rows
#define H1_STRIDE 520                    // bf16 MLP hidden half (512 + pad)

#define XW_OFF 0u                        // fp32 [49][260]  = 50960
#define XN_OFF 50960u                    // bf16 [64][264]  = 33792 (normed / attn-out)
#define Q_OFF  84752u                    // bf16 [64][264]  = 33792 (q / ln2-out)
#define K_OFF  118544u                   // bf16 [64][264]  = 33792
#define VT_OFF 152336u                   // bf16 [256][72]  = 36864
#define P_OFF  189200u                   // bf16 [8][16][72]= 18432
#define SMEM_BYTES 207632u
#define H1_OFF K_OFF                     // bf16 [64][520] = 66560, reuses k/vT

__device__ __forceinline__ int token_cnt(int t, int wy, int wx) {
  int th = t / WSZ, tw = t % WSZ;
  int h = wy * WSZ + th, w = wx * WSZ + tw;
  int rh = (h >= HP - WSZ) + (h >= HP - SSH);
  int rw = (w >= HP - WSZ) + (w >= HP - SSH);
  return 3 * rh + rw;
}

__device__ __forceinline__ v8f wmma_bf16(const Frag& a, const Frag& b, v8f c) {
  return __builtin_amdgcn_wmma_f32_16x16x32_bf16(false, a.h, false, b.h,
                                                 (short)0, c, false, false);
}

// A fragment: row-major matrix row; lane half lh reads K chunks
// {kt*32+lh*8 .. +8} and {kt*32+lh*8+16 .. +8}  (ISA 16-bit A layout)
__device__ __forceinline__ Frag load_a(const __bf16* row, int kt, int lh) {
  Frag f;
  const __bf16* p = row + kt * 32 + lh * 8;
  f.u[0] = *reinterpret_cast<const u32x4*>(p);
  f.u[1] = *reinterpret_cast<const u32x4*>(p + 16);
  return f;
}

// B fragment: matrix stored [N][K] row-major (i.e. weight [out][in]); lane
// half lh reads K chunks {kt*32+lh*8 .. +8} and {kt*32+lh*8+16 .. +8}
__device__ __forceinline__ Frag load_b(const __bf16* row, int kt, int lh) {
  Frag f;
  const __bf16* p = row + kt * 32 + lh * 8;
  f.u[0] = *reinterpret_cast<const u32x4*>(p);
  f.u[1] = *reinterpret_cast<const u32x4*>(p + 16);
  return f;
}

// ---------------------------------------------------------------------------
__global__ void convert_weights_kernel(const float* __restrict__ wqkv,
                                       const float* __restrict__ wo,
                                       const float* __restrict__ w1,
                                       const float* __restrict__ w2,
                                       __bf16* __restrict__ o) {
  int i = blockIdx.x * 256 + threadIdx.x;           // 786432 total
  float v;
  if (i < 196608)       v = wqkv[i];
  else if (i < 262144)  v = wo[i - 196608];
  else if (i < 524288)  v = w1[i - 262144];
  else                  v = w2[i - 524288];
  o[i] = (__bf16)v;
}

// ---------------------------------------------------------------------------
__global__ __launch_bounds__(256) void swin_block_kernel(
    const float* __restrict__ x,
    const float* __restrict__ b_qkv, const float* __restrict__ b_o,
    const float* __restrict__ ln1_g, const float* __restrict__ ln1_b,
    const float* __restrict__ ln2_g, const float* __restrict__ ln2_b,
    const float* __restrict__ b1, const float* __restrict__ b2,
    const __bf16* __restrict__ wqkv_bf, const __bf16* __restrict__ wo_bf,
    const __bf16* __restrict__ w1_bf,   const __bf16* __restrict__ w2_bf,
    float* __restrict__ out) {
  extern __shared__ char smem[];
  float*  xw = reinterpret_cast<float*>(smem + XW_OFF);
  __bf16* xn = reinterpret_cast<__bf16*>(smem + XN_OFF);   // normed -> attn out
  __bf16* qb = reinterpret_cast<__bf16*>(smem + Q_OFF);    // q -> ln2 out
  __bf16* kb = reinterpret_cast<__bf16*>(smem + K_OFF);
  __bf16* vT = reinterpret_cast<__bf16*>(smem + VT_OFF);
  __bf16* pb = reinterpret_cast<__bf16*>(smem + P_OFF);
  __bf16* h1 = reinterpret_cast<__bf16*>(smem + H1_OFF);

  const int tid  = threadIdx.x;
  const int wave = tid >> 5;
  const int lane = tid & 31;
  const int l16  = lane & 15;
  const int lh   = lane >> 4;

  const int bid = blockIdx.x;            // 0 .. 8*256-1
  const int b   = bid >> 8;
  const int w   = bid & 255;
  const int wy  = w >> 4;
  const int wx  = w & 15;

  // ---- 1) load shifted window: xw[t][c] = x[b][c][(wy*7+th+3)%112][(wx*7+tw+3)%112]
  for (int it = 0; it < TT; ++it) {
    int j  = it * 256 + tid;             // (c, th, tw), tw innermost for coalescing
    int tw = j % WSZ;
    int th = (j / WSZ) % WSZ;
    int c  = j / (WSZ * WSZ);
    int gh = (wy * WSZ + th + SSH) % HP;
    int gw = (wx * WSZ + tw + SSH) % HP;
    xw[(th * WSZ + tw) * XW_STRIDE + c] =
        x[(((size_t)b * CC + c) * HP + gh) * HP + gw];
  }
  __syncthreads();

  // ---- 2) LayerNorm1 -> xn (bf16, packed b128 stores), zero pad rows 49..63
  {
    int row = tid >> 2, sub = tid & 3;   // 4 lanes per row, 64 features each
    __bf16* xo = xn + row * XN_STRIDE + sub * 64;
    if (row < TT) {
      const float* xr = xw + row * XW_STRIDE;
      float s = 0.f, s2 = 0.f;
#pragma unroll
      for (int i = 0; i < 64; ++i) { float v = xr[sub * 64 + i]; s += v; s2 += v * v; }
      s  += __shfl_xor(s, 1, 32);  s  += __shfl_xor(s, 2, 32);
      s2 += __shfl_xor(s2, 1, 32); s2 += __shfl_xor(s2, 2, 32);
      float mu  = s * (1.0f / CC);
      float inv = rsqrtf(s2 * (1.0f / CC) - mu * mu + 1e-5f);
#pragma unroll
      for (int i8 = 0; i8 < 8; ++i8) {
        Pack8 pk;
#pragma unroll
        for (int k = 0; k < 8; ++k) {
          int f = sub * 64 + i8 * 8 + k;
          pk.e[k] = (__bf16)((xr[f] - mu) * inv * ln1_g[f] + ln1_b[f]);
        }
        *reinterpret_cast<u32x4*>(xo + i8 * 8) = pk.u;
      }
    } else {
      const u32x4 z4 = {0u, 0u, 0u, 0u};
#pragma unroll
      for (int i8 = 0; i8 < 8; ++i8)
        *reinterpret_cast<u32x4*>(xo + i8 * 8) = z4;
    }
  }
  __syncthreads();

  // ---- 3) QKV GEMM: [64x256] @ [256x768] ; 192 tiles over 8 waves
  for (int ti = wave; ti < 192; ti += 8) {
    int mt = ti & 3, nt = ti >> 2;       // nt 0..47 (uniform per wave-iteration)
    v8f acc = {0.f,0.f,0.f,0.f,0.f,0.f,0.f,0.f};
    const __bf16* arow = xn + (mt * 16 + l16) * XN_STRIDE;
    const __bf16* brow = wqkv_bf + (size_t)(nt * 16 + l16) * CC;
#pragma unroll
    for (int kt = 0; kt < 8; ++kt)
      acc = wmma_bf16(load_a(arow, kt, lh), load_b(brow, kt, lh), acc);
    float bias = b_qkv[nt * 16 + l16];
    if (nt < 16) {                       // Q: token-major, fold 1/sqrt(hd)
      int fcol = nt * 16 + l16;
#pragma unroll
      for (int j = 0; j < 8; ++j)
        qb[(mt * 16 + lh * 8 + j) * XN_STRIDE + fcol] =
            (__bf16)((acc[j] + bias) * 0.17677669529663687f);
    } else if (nt < 32) {                // K: token-major
      int fcol = nt * 16 + l16 - 256;
#pragma unroll
      for (int j = 0; j < 8; ++j)
        kb[(mt * 16 + lh * 8 + j) * XN_STRIDE + fcol] = (__bf16)(acc[j] + bias);
    } else {                             // V: feature-major, contiguous -> b128
      int f = nt * 16 + l16 - 512;
      Pack8 pk;
#pragma unroll
      for (int j = 0; j < 8; ++j) pk.e[j] = (__bf16)(acc[j] + bias);
      *reinterpret_cast<u32x4*>(vT + f * VT_STRIDE + mt * 16 + lh * 8) = pk.u;
    }
  }
  __syncthreads();

  // ---- 4) attention: wave == head; scores (K=32, one WMMA per tile),
  //        masked softmax in registers, P @ V (K=64, two WMMA)
  {
    const int h = wave;
    __bf16* ph = pb + h * 16 * P_STRIDE;
    for (int mt = 0; mt < 4; ++mt) {
      Frag fq = load_a(qb + (mt * 16 + l16) * XN_STRIDE + h * HD, 0, lh);
      v8f sc[4];
#pragma unroll
      for (int nt2 = 0; nt2 < 4; ++nt2) {
        v8f z = {0.f,0.f,0.f,0.f,0.f,0.f,0.f,0.f};
        Frag fb = load_b(kb + (nt2 * 16 + l16) * XN_STRIDE + h * HD, 0, lh);
        sc[nt2] = wmma_bf16(fq, fb, z);
      }
      int cntc[4]; bool cval[4];
#pragma unroll
      for (int nt2 = 0; nt2 < 4; ++nt2) {
        int c = nt2 * 16 + l16;
        cval[nt2] = (c < TT);
        cntc[nt2] = token_cnt(c, wy, wx);
      }
#pragma unroll
      for (int j = 0; j < 8; ++j) {
        int r  = mt * 16 + lh * 8 + j;
        int cr = token_cnt(r, wy, wx);
        float mx = -3.0e38f;
#pragma unroll
        for (int nt2 = 0; nt2 < 4; ++nt2) {
          float v = sc[nt2][j];
          if (!cval[nt2] || cntc[nt2] != cr) v = -1.0e30f;   // shift mask + col pad
          sc[nt2][j] = v;
          mx = fmaxf(mx, v);
        }
        mx = fmaxf(mx, __shfl_xor(mx, 1, 32)); mx = fmaxf(mx, __shfl_xor(mx, 2, 32));
        mx = fmaxf(mx, __shfl_xor(mx, 4, 32)); mx = fmaxf(mx, __shfl_xor(mx, 8, 32));
        float s = 0.f;
#pragma unroll
        for (int nt2 = 0; nt2 < 4; ++nt2) {
          float e = __expf(sc[nt2][j] - mx);
          sc[nt2][j] = e; s += e;
        }
        s += __shfl_xor(s, 1, 32); s += __shfl_xor(s, 2, 32);
        s += __shfl_xor(s, 4, 32); s += __shfl_xor(s, 8, 32);
        float inv = (s > 0.f) ? (1.0f / s) : 0.f;
        __bf16* prow = ph + (lh * 8 + j) * P_STRIDE;
#pragma unroll
        for (int nt2 = 0; nt2 < 4; ++nt2)
          prow[nt2 * 16 + l16] = (__bf16)(sc[nt2][j] * inv);
      }
      // P[16x64] @ V[64x32] -> attn-out tile, written into xn (reused)
      const __bf16* arow = ph + l16 * P_STRIDE;
      Frag pa0 = load_a(arow, 0, lh);
      Frag pa1 = load_a(arow, 1, lh);
#pragma unroll
      for (int nt = 0; nt < 2; ++nt) {
        v8f o = {0.f,0.f,0.f,0.f,0.f,0.f,0.f,0.f};
        const __bf16* brow = vT + (h * HD + nt * 16 + l16) * VT_STRIDE;
        o = wmma_bf16(pa0, load_b(brow, 0, lh), o);
        o = wmma_bf16(pa1, load_b(brow, 1, lh), o);
#pragma unroll
        for (int j = 0; j < 8; ++j)
          xn[(mt * 16 + lh * 8 + j) * XN_STRIDE + h * HD + nt * 16 + l16] = (__bf16)o[j];
      }
    }
  }
  __syncthreads();

  // ---- 5) output projection + residual into xw (fp32)
  for (int ti = wave; ti < 64; ti += 8) {
    int mt = ti & 3, nt = ti >> 2;
    v8f acc = {0.f,0.f,0.f,0.f,0.f,0.f,0.f,0.f};
    const __bf16* arow = xn + (mt * 16 + l16) * XN_STRIDE;
    const __bf16* brow = wo_bf + (size_t)(nt * 16 + l16) * CC;
#pragma unroll
    for (int kt = 0; kt < 8; ++kt)
      acc = wmma_bf16(load_a(arow, kt, lh), load_b(brow, kt, lh), acc);
    int   c    = nt * 16 + l16;
    float bias = b_o[c];
#pragma unroll
    for (int j = 0; j < 8; ++j) {
      int r = mt * 16 + lh * 8 + j;
      if (r < TT) xw[r * XW_STRIDE + c] += acc[j] + bias;
    }
  }
  __syncthreads();

  // ---- 6) LayerNorm2 -> qb (reused as MLP input), zero pad rows
  {
    int row = tid >> 2, sub = tid & 3;
    __bf16* xo = qb + row * XN_STRIDE + sub * 64;
    if (row < TT) {
      const float* xr = xw + row * XW_STRIDE;
      float s = 0.f, s2 = 0.f;
#pragma unroll
      for (int i = 0; i < 64; ++i) { float v = xr[sub * 64 + i]; s += v; s2 += v * v; }
      s  += __shfl_xor(s, 1, 32);  s  += __shfl_xor(s, 2, 32);
      s2 += __shfl_xor(s2, 1, 32); s2 += __shfl_xor(s2, 2, 32);
      float mu  = s * (1.0f / CC);
      float inv = rsqrtf(s2 * (1.0f / CC) - mu * mu + 1e-5f);
#pragma unroll
      for (int i8 = 0; i8 < 8; ++i8) {
        Pack8 pk;
#pragma unroll
        for (int k = 0; k < 8; ++k) {
          int f = sub * 64 + i8 * 8 + k;
          pk.e[k] = (__bf16)((xr[f] - mu) * inv * ln2_g[f] + ln2_b[f]);
        }
        *reinterpret_cast<u32x4*>(xo + i8 * 8) = pk.u;
      }
    } else {
      const u32x4 z4 = {0u, 0u, 0u, 0u};
#pragma unroll
      for (int i8 = 0; i8 < 8; ++i8)
        *reinterpret_cast<u32x4*>(xo + i8 * 8) = z4;
    }
  }
  __syncthreads();

  // ---- 7) MLP in two 512-wide halves; final [64x256] tiles accumulate in regs
  v8f yacc[8];
#pragma unroll
  for (int i = 0; i < 8; ++i) yacc[i] = (v8f){0.f,0.f,0.f,0.f,0.f,0.f,0.f,0.f};

  for (int hv = 0; hv < 2; ++hv) {
    // 7a) h1_half = gelu(xn2 @ w1_half^T + b1_half)   (128 tiles / 8 waves)
    for (int ti = wave; ti < 128; ti += 8) {
      int mt = ti & 3, nt = ti >> 2;               // nt 0..31
      int fo = hv * 512 + nt * 16 + l16;
      v8f acc = {0.f,0.f,0.f,0.f,0.f,0.f,0.f,0.f};
      const __bf16* arow = qb + (mt * 16 + l16) * XN_STRIDE;
      const __bf16* brow = w1_bf + (size_t)fo * CC;
#pragma unroll
      for (int kt = 0; kt < 8; ++kt)
        acc = wmma_bf16(load_a(arow, kt, lh), load_b(brow, kt, lh), acc);
      float bias = b1[fo];
#pragma unroll
      for (int j = 0; j < 8; ++j) {
        int r = mt * 16 + lh * 8 + j;
        float v = acc[j] + bias;
        float g = 0.5f * v * (1.0f + erff(v * 0.70710678118654752f)); // exact gelu
        h1[r * H1_STRIDE + nt * 16 + l16] = (__bf16)g;
      }
    }
    __syncthreads();
    // 7b) y += h1_half @ w2_half^T   (8 persistent tiles per wave, K=512)
#pragma unroll
    for (int i = 0; i < 8; ++i) {
      int ti = wave + i * 8;
      int mt = ti & 3, nt = ti >> 2;
      const __bf16* arow = h1 + (mt * 16 + l16) * H1_STRIDE;
      const __bf16* brow = w2_bf + (size_t)(nt * 16 + l16) * 1024 + hv * 512;
#pragma unroll
      for (int kt = 0; kt < 16; ++kt)
        yacc[i] = wmma_bf16(load_a(arow, kt, lh), load_b(brow, kt, lh), yacc[i]);
    }
    __syncthreads();
  }

  // ---- 8) + b2 + residual, write back with inverse shift
#pragma unroll
  for (int i = 0; i < 8; ++i) {
    int ti = wave + i * 8;
    int mt = ti & 3, nt = ti >> 2;
    int c = nt * 16 + l16;
    float bias = b2[c];
#pragma unroll
    for (int j = 0; j < 8; ++j) {
      int r = mt * 16 + lh * 8 + j;
      if (r < TT) {
        float v = yacc[i][j] + bias + xw[r * XW_STRIDE + c];
        int th = r / WSZ, tw = r % WSZ;
        int gh = (wy * WSZ + th + SSH) % HP;
        int gw = (wx * WSZ + tw + SSH) % HP;
        out[(((size_t)b * CC + c) * HP + gh) * HP + gw] = v;
      }
    }
  }
}

// ---------------------------------------------------------------------------
extern "C" void kernel_launch(void* const* d_in, const int* in_sizes, int n_in,
                              void* d_out, int out_size, void* d_ws, size_t ws_size,
                              hipStream_t stream) {
  (void)in_sizes; (void)n_in; (void)out_size; (void)ws_size;
  const float* x     = (const float*)d_in[0];
  const float* w_qkv = (const float*)d_in[1];
  const float* b_qkv = (const float*)d_in[2];
  const float* w_o   = (const float*)d_in[3];
  const float* b_o   = (const float*)d_in[4];
  const float* ln1_g = (const float*)d_in[5];
  const float* ln1_b = (const float*)d_in[6];
  const float* ln2_g = (const float*)d_in[7];
  const float* ln2_b = (const float*)d_in[8];
  const float* w1    = (const float*)d_in[9];
  const float* b1    = (const float*)d_in[10];
  const float* w2    = (const float*)d_in[11];
  const float* b2    = (const float*)d_in[12];

  __bf16* wsb = (__bf16*)d_ws;                 // 786432 bf16 = 1.5 MB
  convert_weights_kernel<<<3072, 256, 0, stream>>>(w_qkv, w_o, w1, w2, wsb);
  swin_block_kernel<<<8 * 256, 256, SMEM_BYTES, stream>>>(
      x, b_qkv, b_o, ln1_g, ln1_b, ln2_g, ln2_b, b1, b2,
      wsb, wsb + 196608, wsb + 262144, wsb + 524288, (float*)d_out);
}
